// Attention_28200755265553
// MI455X (gfx1250) — matrix operand order repaired and checked by reading, not compiled, not run on hardware
//
#include <hip/hip_runtime.h>
#include <hip/hip_bf16.h>
#include <math.h>

// ---------------------------------------------------------------------------
// Luong "general" attention, B=64, T=128, S=1024, I=O=1024, C=I+O=2048
// Outputs (concat): attn_outputs [B,T,O] f32, attn_weights [B,T,S] f32
// Workspace: q_proj f16 (16MB) | probs f16 (16MB) | context f16 (16MB)
// ---------------------------------------------------------------------------

#define BB   64
#define TT   128
#define SS   1024
#define II   1024
#define OO   1024
#define CC   2048

typedef __attribute__((ext_vector_type(16))) _Float16 v16h;
typedef __attribute__((ext_vector_type(8)))  _Float16 v8h;
typedef __attribute__((ext_vector_type(8)))  float    v8f;
typedef __attribute__((ext_vector_type(4)))  float    f4;

// ---- WMMA fragment loaders (wave32 layouts per CDNA5 ISA 7.12.2) -----------
// A 16x32 f16 (MxK), row-major source, ld in elements:
//   lanes 0-15: M=lane,    elems 0-7 -> K=0..7,  8-15 -> K=16..23
//   lanes16-31: M=lane-16, elems 0-7 -> K=8..15, 8-15 -> K=24..31
__device__ inline v16h load_a_f32(const float* __restrict__ p, int ld, int lane) {
    int row = lane & 15, kb = (lane >> 4) << 3;
    const float* q = p + (size_t)row * ld + kb;
    f4 u0 = *(const f4*)(q);
    f4 u1 = *(const f4*)(q + 4);
    f4 u2 = *(const f4*)(q + 16);
    f4 u3 = *(const f4*)(q + 20);
    v16h a;
#pragma unroll
    for (int j = 0; j < 4; ++j) {
        a[j]      = (_Float16)u0[j];
        a[j + 4]  = (_Float16)u1[j];
        a[j + 8]  = (_Float16)u2[j];
        a[j + 12] = (_Float16)u3[j];
    }
    return a;
}

__device__ inline v16h load_a_f16(const _Float16* __restrict__ p, int ld, int lane) {
    int row = lane & 15, kb = (lane >> 4) << 3;
    const _Float16* q = p + (size_t)row * ld + kb;
    v8h u0 = *(const v8h*)(q);
    v8h u1 = *(const v8h*)(q + 16);
    v16h a;
#pragma unroll
    for (int j = 0; j < 8; ++j) { a[j] = u0[j]; a[j + 8] = u1[j]; }
    return a;
}

// B 32x16 (KxN) with B[k][n] = M[n][k] (GEMM vs B^T; M row-major, ld):
//   lanes 0-15: N=lane, elems 0-7 -> K=0..7, 8-15 -> K=16..23; lanes16-31: N=lane-16, K=8..15, 24..31
__device__ inline v16h load_bt_f32(const float* __restrict__ p, int ld, int lane) {
    int n = lane & 15, kb = (lane >> 4) << 3;
    const float* q = p + (size_t)n * ld + kb;
    f4 u0 = *(const f4*)(q);
    f4 u1 = *(const f4*)(q + 4);
    f4 u2 = *(const f4*)(q + 16);
    f4 u3 = *(const f4*)(q + 20);
    v16h b;
#pragma unroll
    for (int j = 0; j < 4; ++j) {
        b[j]      = (_Float16)u0[j];
        b[j + 4]  = (_Float16)u1[j];
        b[j + 8]  = (_Float16)u2[j];
        b[j + 12] = (_Float16)u3[j];
    }
    return b;
}

__device__ inline v16h load_bt_f16(const _Float16* __restrict__ p, int ld, int lane) {
    int n = lane & 15, kb = (lane >> 4) << 3;
    const _Float16* q = p + (size_t)n * ld + kb;
    v8h u0 = *(const v8h*)(q);
    v8h u1 = *(const v8h*)(q + 16);
    v16h b;
#pragma unroll
    for (int j = 0; j < 8; ++j) { b[j] = u0[j]; b[j + 8] = u1[j]; }
    return b;
}

// C/D 16x16 f32: col N = lane&15; VGPR r -> row M = r + (lane>=16 ? 8 : 0)
__device__ inline void store_tile_f16(_Float16* __restrict__ dst, int ld, int lane, v8f acc) {
    int col = lane & 15, rb = (lane >> 4) << 3;
#pragma unroll
    for (int r = 0; r < 8; ++r) dst[(size_t)(rb + r) * ld + col] = (_Float16)acc[r];
}

#define WMMA(a, b, c) \
    __builtin_amdgcn_wmma_f32_16x16x32_f16(false, (a), false, (b), (short)0, (c), false, false)

// ---------------------------------------------------------------------------
// Kernel 1: q_proj = dec @ W_attn^T  -> f16
// grid = B * (T/16) * (I/128); 8 waves, one 16x16 tile each
// ---------------------------------------------------------------------------
__global__ void __launch_bounds__(256)
qproj_kernel(const float* __restrict__ dec, const float* __restrict__ Wattn,
             _Float16* __restrict__ qh) {
    int bid   = blockIdx.x;
    int iBlk  = bid & 7;
    int tTile = (bid >> 3) & 7;
    int b     = bid >> 6;
    int wave  = threadIdx.x >> 5;
    int lane  = threadIdx.x & 31;
    int i0 = iBlk * 128 + wave * 16;
    int t0 = tTile * 16;

    const float* A  = dec   + ((size_t)b * TT + t0) * OO;
    const float* Bm = Wattn + (size_t)i0 * OO;

    v8f acc = {};
    for (int kc = 0; kc < OO; kc += 32) {
        v16h a  = load_a_f32(A + kc, OO, lane);
        v16h bf = load_bt_f32(Bm + kc, OO, lane);
        acc = WMMA(a, bf, acc);
    }
    store_tile_f16(qh + ((size_t)b * TT + t0) * II + i0, II, lane, acc);
}

// ---------------------------------------------------------------------------
// Kernel 2: scores = q @ enc^T, mask, softmax -> f32 weights + f16 probs
// grid = B * (T/16); block computes 16 x S scores in 64KB LDS.
// Wave owns 8 N-tiles (wave*128..+127) and 2 softmax rows.
// ---------------------------------------------------------------------------
__global__ void __launch_bounds__(256)
scores_softmax_kernel(const _Float16* __restrict__ qh, const float* __restrict__ enc,
                      const unsigned char* __restrict__ mask,
                      float* __restrict__ wts, _Float16* __restrict__ ph) {
    __shared__ float sc[16 * SS];                      // 64 KB

    int b    = blockIdx.x >> 3;
    int t0   = (blockIdx.x & 7) * 16;
    int wave = threadIdx.x >> 5;
    int lane = threadIdx.x & 31;

    const _Float16* Q = qh  + ((size_t)b * TT + t0) * II;
    const float*    E = enc + (size_t)b * SS * II;

    v8f acc[8];
#pragma unroll
    for (int nt = 0; nt < 8; ++nt) acc[nt] = (v8f){};

    for (int kc = 0; kc < II; kc += 32) {
        v16h a = load_a_f16(Q + kc, II, lane);
#pragma unroll
        for (int nt = 0; nt < 8; ++nt) {
            int s0 = wave * 128 + nt * 16;
            v16h bf = load_bt_f32(E + (size_t)s0 * II + kc, II, lane);
            acc[nt] = WMMA(a, bf, acc[nt]);
        }
    }

    int col = lane & 15, rb = (lane >> 4) << 3;
#pragma unroll
    for (int nt = 0; nt < 8; ++nt) {
        int s0 = wave * 128 + nt * 16;
#pragma unroll
        for (int r = 0; r < 8; ++r) sc[(rb + r) * SS + s0 + col] = acc[nt][r];
    }
    __syncthreads();

    // masked softmax; wave handles rows 2*wave, 2*wave+1 (wave32 shuffles)
    const unsigned char* M = mask + (size_t)b * SS;
    for (int rr = 0; rr < 2; ++rr) {
        int row = wave * 2 + rr;
        float* srow = sc + row * SS;

        float mx = -INFINITY;
        for (int s = lane; s < SS; s += 32) {
            float v = M[s] ? -INFINITY : srow[s];
            srow[s] = v;
            mx = fmaxf(mx, v);
        }
#pragma unroll
        for (int off = 16; off > 0; off >>= 1) mx = fmaxf(mx, __shfl_xor(mx, off, 32));

        float sum = 0.0f;
        for (int s = lane; s < SS; s += 32) {
            float e = __expf(srow[s] - mx);
            srow[s] = e;
            sum += e;
        }
#pragma unroll
        for (int off = 16; off > 0; off >>= 1) sum += __shfl_xor(sum, off, 32);

        float inv = 1.0f / sum;
        size_t rowOff = ((size_t)b * TT + t0 + row) * SS;
        for (int s = lane; s < SS; s += 32) {
            float w = srow[s] * inv;
            wts[rowOff + s] = w;
            ph[rowOff + s]  = (_Float16)w;
        }
    }
}

// ---------------------------------------------------------------------------
// Kernel 3: context = probs @ enc (A·B).
// enc slab staged TRANSPOSED into LDS as f16: eTt[col][k], row stride 40
// halves (80 B) -> 16B-aligned rows, conflict-free ds_load_b128 fragments.
// grid = B * (T/16) * (I/128)
// ---------------------------------------------------------------------------
__global__ void __launch_bounds__(256)
context_kernel(const _Float16* __restrict__ ph, const float* __restrict__ enc,
               _Float16* __restrict__ ctx) {
    __shared__ _Float16 eTt[128][40];                  // 10 KB, padded

    int bid   = blockIdx.x;
    int iBlk  = bid & 7;
    int tTile = (bid >> 3) & 7;
    int b     = bid >> 6;
    int wave  = threadIdx.x >> 5;
    int lane  = threadIdx.x & 31;
    int i0 = iBlk * 128;
    int t0 = tTile * 16;

    const _Float16* P = ph  + ((size_t)b * TT + t0) * SS;
    const float*    E = enc + (size_t)b * SS * II + i0;

    v8f acc = {};
    for (int kc = 0; kc < SS; kc += 32) {
        // stage enc[kc..kc+32)[i0..i0+128) -> eTt[col][k] (f16, transposed)
        for (int x = threadIdx.x; x < 32 * 32; x += 256) {
            int rr = x >> 5;            // k row 0..31
            int cg = (x & 31) << 2;     // col group 0,4,...,124
            f4 v = *(const f4*)(E + (size_t)(kc + rr) * II + cg);
#pragma unroll
            for (int j = 0; j < 4; ++j) eTt[cg + j][rr] = (_Float16)v[j];
        }
        __syncthreads();
        v16h a  = load_a_f16(P + kc, SS, lane);
        v16h bf = load_bt_f16(&eTt[wave * 16][0], 40, lane);  // B[k][n]=eTt[n][k]
        acc = WMMA(a, bf, acc);
        __syncthreads();
    }
    store_tile_f16(ctx + ((size_t)b * TT + t0) * II + i0 + wave * 16, II, lane, acc);
}

// ---------------------------------------------------------------------------
// Kernel 4: out = tanh( [ctx|dec] @ W_out^T + b_out )
// grid = B * (T/16) * (O/128)
// ---------------------------------------------------------------------------
__global__ void __launch_bounds__(256)
outproj_kernel(const _Float16* __restrict__ ctx, const float* __restrict__ dec,
               const float* __restrict__ Wout, const float* __restrict__ bout,
               float* __restrict__ out) {
    int bid   = blockIdx.x;
    int oBlk  = bid & 7;
    int tTile = (bid >> 3) & 7;
    int b     = bid >> 6;
    int wave  = threadIdx.x >> 5;
    int lane  = threadIdx.x & 31;
    int o0 = oBlk * 128 + wave * 16;
    int t0 = tTile * 16;

    const _Float16* C = ctx + ((size_t)b * TT + t0) * II;
    const float*    D = dec + ((size_t)b * TT + t0) * OO;
    const float*    W = Wout + (size_t)o0 * CC;

    v8f acc = {};
    for (int kc = 0; kc < II; kc += 32) {             // context half of concat
        v16h a  = load_a_f16(C + kc, II, lane);
        v16h bf = load_bt_f32(W + kc, CC, lane);
        acc = WMMA(a, bf, acc);
    }
    for (int kc = 0; kc < OO; kc += 32) {             // decoder half of concat
        v16h a  = load_a_f32(D + kc, OO, lane);
        v16h bf = load_bt_f32(W + II + kc, CC, lane);
        acc = WMMA(a, bf, acc);
    }

    int col = lane & 15, rb = (lane >> 4) << 3;
    float bo = bout[o0 + col];
#pragma unroll
    for (int r = 0; r < 8; ++r)
        out[((size_t)b * TT + t0 + rb + r) * OO + o0 + col] = tanhf(acc[r] + bo);
}

// ---------------------------------------------------------------------------
extern "C" void kernel_launch(void* const* d_in, const int* in_sizes, int n_in,
                              void* d_out, int out_size, void* d_ws, size_t ws_size,
                              hipStream_t stream) {
    const float*         dec   = (const float*)d_in[0];         // [B,T,O]
    const float*         enc   = (const float*)d_in[1];         // [B,S,I]
    const unsigned char* mask  = (const unsigned char*)d_in[2]; // [B,S] bool
    const float*         Wattn = (const float*)d_in[3];         // [I,O]
    const float*         Wout  = (const float*)d_in[4];         // [O,C]
    const float*         bout  = (const float*)d_in[5];         // [O]

    float* attn_out = (float*)d_out;                            // [B,T,O]
    float* wts      = attn_out + (size_t)BB * TT * OO;          // [B,T,S]

    _Float16* qh  = (_Float16*)d_ws;                            // [B,T,I]
    _Float16* ph  = qh + (size_t)BB * TT * II;                  // [B,T,S]
    _Float16* ctx = ph + (size_t)BB * TT * SS;                  // [B,T,I]

    qproj_kernel<<<BB * (TT / 16) * (II / 128), 256, 0, stream>>>(dec, Wattn, qh);
    scores_softmax_kernel<<<BB * (TT / 16), 256, 0, stream>>>(qh, enc, mask, wts, ph);
    context_kernel<<<BB * (TT / 16) * (II / 128), 256, 0, stream>>>(ph, enc, ctx);
    outproj_kernel<<<BB * (TT / 16) * (OO / 128), 256, 0, stream>>>(ctx, dec, Wout, bout, attn_out);
}
